// CausalSelfAttention_6648609374286
// MI455X (gfx1250) — matrix operand order repaired and checked by reading, not compiled, not run on hardware
//
#include <hip/hip_runtime.h>
#include <stdint.h>
#include <stddef.h>

// ---------------------------------------------------------------------------
// CDNA5 (gfx1250) causal self-attention forward, bf16 WMMA everywhere.
//   v_wmma_f32_16x16x32_bf16:  D(16x16 f32) = A(16x32 bf16) x B(32x16 bf16) + C
// Compute-bound (~70 GFLOP over ~150 MB -> AI ~500 FLOP/B, far past the HBM
// ridge at 23.3 TB/s) -> matrix pipes in bf16, fp32 accumulation.
// GEMMs stage the block-shared operand slab in LDS via the gfx1250 async
// copy path (global_load_async_to_lds_b128 / s_wait_asynccnt), double
// buffered, removing the 4x redundant L2 fetch of the shared slab.
// ---------------------------------------------------------------------------

typedef __bf16 bf16_t;
typedef __attribute__((ext_vector_type(8)))  bf16_t v8bf;
typedef __attribute__((ext_vector_type(16))) bf16_t v16bf;
typedef __attribute__((ext_vector_type(8)))  float  v8f;

#define D_MODEL  1024
#define N_HEADS  16
#define HEAD_DIM 64
#define BATCH    2
#define SEQ      2048
#define MROWS    (BATCH * SEQ)   // 4096

#define NEG_INF (-__builtin_inff())
// attention scale folded with log2(e): softmax computed as exp2(s' - m')
#define QSCALE   0.1803368801111204f   // 0.125 * log2(e)

// native bf16 convert (RNE) -> v_cvt_pk_bf16_f32 when vectorized
__device__ __forceinline__ bf16_t f2bf(float f) { return (bf16_t)f; }

__device__ __forceinline__ v8f wmma_bf16(v16bf a, v16bf b, v8f c) {
  // (neg_a, A, neg_b, B, c_mod, C, reuse_a, reuse_b)
  return __builtin_amdgcn_wmma_f32_16x16x32_bf16(false, a, false, b,
                                                 (short)0, c, false, false);
}

// A-matrix fragment (16x32 bf16), contiguous k order:
//   lanes 0-15 : row M = lane,    K = 0..15
//   lanes 16-31: row M = lane-16, K = 16..31
__device__ __forceinline__ v16bf load_frag_a(const bf16_t* row_k0, int hi8) {
  v8bf lo = *(const v8bf*)(row_k0 + 2 * hi8);
  v8bf hi = *(const v8bf*)(row_k0 + 2 * hi8 + 8);
  return __builtin_shufflevector(lo, hi, 0, 1, 2, 3, 4, 5, 6, 7,
                                 8, 9, 10, 11, 12, 13, 14, 15);
}
// B-matrix fragment (32x16 bf16): lane = column N; lanes 0-15 hold K=0..15
// contiguous, lanes 16-31 hold K=16..31.

// ---------------------------------------------------------------------------
// Async slab staging: 32 rows x 32 k (64 B/row) of a [*, D_MODEL] bf16 matrix
// into LDS, rows padded to 80 B so the 16 B fragment reads spread over banks.
// One global_load_async_to_lds_b128 per thread (128 threads = 2 KB slab).
// Tracked by ASYNCcnt -> s_wait_asynccnt before the workgroup barrier.
// ---------------------------------------------------------------------------
#define SLAB_STRIDE 80
#define SLAB_BYTES  (32 * SLAB_STRIDE)

__device__ __forceinline__ void slab_async_load(char* lds_buf,
                                                const bf16_t* src, int row0,
                                                int k, int tid) {
  const int row  = tid >> 2;  // 0..31
  const int quad = tid & 3;   // 16 B each
  const bf16_t* g = src + (size_t)(row0 + row) * D_MODEL + k + quad * 8;
  // low 32 bits of the flat address of a __shared__ object = LDS byte offset
  uint32_t lds_off = (uint32_t)(uintptr_t)(lds_buf + row * SLAB_STRIDE + quad * 16);
  asm volatile("global_load_async_to_lds_b128 %0, %1, off"
               :
               : "v"(lds_off), "v"(g)
               : "memory");
}

__device__ __forceinline__ void wait_async0() {
  asm volatile("s_wait_asynccnt 0x0" ::: "memory");
}

// read one B-fragment (lane = column m/n, 32 B of one slab row) from LDS
__device__ __forceinline__ v16bf slab_frag_b(const char* lds_buf, int row,
                                             int hi32b) {
  const char* rp = lds_buf + row * SLAB_STRIDE + hi32b;
  v8bf lo = *(const v8bf*)rp;
  v8bf hi = *(const v8bf*)(rp + 16);
  return __builtin_shufflevector(lo, hi, 0, 1, 2, 3, 4, 5, 6, 7,
                                 8, 9, 10, 11, 12, 13, 14, 15);
}

// ---------------------------------------------------------------------------
// fp32 -> bf16 elementwise conversion
// ---------------------------------------------------------------------------
__global__ __launch_bounds__(256) void f32_to_bf16_kernel(
    const float* __restrict__ in, bf16_t* __restrict__ out, int n) {
  int i = blockIdx.x * blockDim.x + threadIdx.x;
  int stride = gridDim.x * blockDim.x;
  for (; i < n; i += stride) out[i] = f2bf(in[i]);
}

// ---------------------------------------------------------------------------
// GEMM, transposed orientation:  Ot[n, m] = (sum_k W[n,k] * A[m,k] + bias[n]) * cscale
//   wmma A-frag = W rows (M-dim = n, per-wave exclusive, global loads)
//   wmma B-frag = A^T   (N-dim = m, block-shared -> async LDS slab)
//   C layout: lane -> m, VGPR r -> n contiguous -> packed stores.
// OUT_MODE 0: bf16 into [B,H,T,Dh] (Q/K).  OUT_MODE 2: fp32 [m,n] (final).
// Wave tile 64n x 32m (4x2 WMMA). Block = 4 waves = 256 n.
// grid = (MROWS/32, D_MODEL/256)
// ---------------------------------------------------------------------------
template <int OUT_MODE>
__global__ __launch_bounds__(128) void gemm_xwT_t(
    const bf16_t* __restrict__ A,   // [MROWS, D_MODEL]
    const bf16_t* __restrict__ W,   // [D_MODEL, D_MODEL]
    const float* __restrict__ bias, // [D_MODEL]
    float cscale, void* __restrict__ outv) {
  __shared__ __attribute__((aligned(16))) char lds_slab[2][SLAB_BYTES];
  const int tid  = threadIdx.x;
  const int lane = tid & 31;
  const int wave = tid >> 5;
  const int ln   = lane & 15;
  const int hi8  = (lane >= 16) ? 8 : 0;
  const int hi32b = (lane >= 16) ? 32 : 0;
  const int m0 = blockIdx.x * 32;
  const int n0 = blockIdx.y * 256 + wave * 64;

  v8f c[4][2];
#pragma unroll
  for (int i = 0; i < 4; ++i)
#pragma unroll
    for (int j = 0; j < 2; ++j) c[i][j] = {};

  slab_async_load(lds_slab[0], A, m0, 0, tid);
  wait_async0();
  __syncthreads();

  int buf = 0;
  for (int k = 0; k < D_MODEL; k += 32) {
    if (k + 32 < D_MODEL)
      slab_async_load(lds_slab[buf ^ 1], A, m0, k + 32, tid);

    v16bf bfr[2];
#pragma unroll
    for (int j = 0; j < 2; ++j)
      bfr[j] = slab_frag_b(lds_slab[buf], j * 16 + ln, hi32b);

#pragma unroll
    for (int i = 0; i < 4; ++i) {
      const int n = n0 + i * 16 + ln;
      const bf16_t* wp = W + (size_t)n * D_MODEL + k;
      __builtin_prefetch(wp + 64, 0, 1);  // k-block after next (k + 64) -> global_prefetch_b8
      v16bf af = load_frag_a(wp, hi8);
#pragma unroll
      for (int j = 0; j < 2; ++j) c[i][j] = wmma_bf16(af, bfr[j], c[i][j]);
    }
    wait_async0();   // next slab resident (ASYNCcnt not covered by barrier)
    __syncthreads(); // everyone done reading current slab
    buf ^= 1;
  }

  if (OUT_MODE == 0) {
    bf16_t* out = (bf16_t*)outv;
#pragma unroll
    for (int i = 0; i < 4; ++i) {
      const int nb = n0 + i * 16 + hi8;  // 8-aligned, within one head
      const v8f bs = *(const v8f*)(bias + nb);
      const int h = nb >> 6, d = nb & 63;
#pragma unroll
      for (int j = 0; j < 2; ++j) {
        const int m = m0 + j * 16 + ln;
        const int b = m >> 11, t = m & (SEQ - 1);
        v8bf pk;
#pragma unroll
        for (int e = 0; e < 8; ++e) pk[e] = f2bf((c[i][j][e] + bs[e]) * cscale);
        *(v8bf*)(out + ((size_t)((b * N_HEADS + h) * SEQ + t)) * HEAD_DIM + d) = pk;
      }
    }
  } else {
    float* out = (float*)outv;
#pragma unroll
    for (int i = 0; i < 4; ++i) {
      const int nb = n0 + i * 16 + hi8;
      const v8f bs = *(const v8f*)(bias + nb);
#pragma unroll
      for (int j = 0; j < 2; ++j) {
        const int m = m0 + j * 16 + ln;
        *(v8f*)(out + (size_t)m * D_MODEL + nb) = c[i][j] + bs;
      }
    }
  }
}

// ---------------------------------------------------------------------------
// GEMM, normal orientation, for V -> stores V^T as [B,H,Dh,T].
//   wmma A-frag = x rows (per-wave exclusive, global), B-frag = W^T slab
//   (block-shared -> async LDS slab). C: lane -> n(=d), r -> m(=t) contiguous.
// Wave tile 64m x 32n. grid = (MROWS/256, D_MODEL/32)
// ---------------------------------------------------------------------------
__global__ __launch_bounds__(128) void gemm_xwT_vT(
    const bf16_t* __restrict__ A, const bf16_t* __restrict__ W,
    const float* __restrict__ bias, bf16_t* __restrict__ out) {
  __shared__ __attribute__((aligned(16))) char lds_slab[2][SLAB_BYTES];
  const int tid  = threadIdx.x;
  const int lane = tid & 31;
  const int wave = tid >> 5;
  const int ln   = lane & 15;
  const int hi8  = (lane >= 16) ? 8 : 0;
  const int hi32b = (lane >= 16) ? 32 : 0;
  const int m0 = blockIdx.x * 256 + wave * 64;
  const int n0 = blockIdx.y * 32;

  v8f c[4][2];
#pragma unroll
  for (int i = 0; i < 4; ++i)
#pragma unroll
    for (int j = 0; j < 2; ++j) c[i][j] = {};

  slab_async_load(lds_slab[0], W, n0, 0, tid);
  wait_async0();
  __syncthreads();

  int buf = 0;
  for (int k = 0; k < D_MODEL; k += 32) {
    if (k + 32 < D_MODEL)
      slab_async_load(lds_slab[buf ^ 1], W, n0, k + 32, tid);

    v16bf bfr[2];
#pragma unroll
    for (int j = 0; j < 2; ++j)
      bfr[j] = slab_frag_b(lds_slab[buf], j * 16 + ln, hi32b);

#pragma unroll
    for (int i = 0; i < 4; ++i) {
      const int m = m0 + i * 16 + ln;
      const bf16_t* ap = A + (size_t)m * D_MODEL + k;
      __builtin_prefetch(ap + 64, 0, 1);
      v16bf af = load_frag_a(ap, hi8);
#pragma unroll
      for (int j = 0; j < 2; ++j) c[i][j] = wmma_bf16(af, bfr[j], c[i][j]);
    }
    wait_async0();
    __syncthreads();
    buf ^= 1;
  }

#pragma unroll
  for (int j = 0; j < 2; ++j) {
    const int n = n0 + j * 16 + ln;
    const float bv = bias[n];
    const int h = n >> 6, d = n & 63;
#pragma unroll
    for (int i = 0; i < 4; ++i) {
      const int mb = m0 + i * 16 + hi8;  // 8-aligned, same batch
      const int b = mb >> 11, t = mb & (SEQ - 1);
      v8bf pk;
#pragma unroll
      for (int e = 0; e < 8; ++e) pk[e] = f2bf(c[i][j][e] + bv);
      *(v8bf*)(out + ((size_t)((b * N_HEADS + h) * HEAD_DIM + d)) * SEQ + t) = pk;
    }
  }
}

// ---------------------------------------------------------------------------
// Flash attention (causal), one wave per (b, h, 16-query tile).
// S^T = K x Q^T so queries sit on the lane axis: softmax stats are per-lane
// scalars; a single half-swap shuffle rebuilds P^T as a B-fragment for
// O^T = V^T x P^T.  Q is pre-scaled by 0.125*log2(e), softmax uses exp2
// (v_exp_f32 is natively base-2).
// grid = (SEQ/16, BATCH*N_HEADS), block = 32 (1 wave).
// ---------------------------------------------------------------------------
__global__ __launch_bounds__(32) void flash_attn_kernel(
    const bf16_t* __restrict__ Q,   // [B,H,T,Dh], pre-scaled
    const bf16_t* __restrict__ K,   // [B,H,T,Dh]
    const bf16_t* __restrict__ Vt,  // [B,H,Dh,T]
    bf16_t* __restrict__ Y) {       // [B,T,D_MODEL]
  const int lane  = threadIdx.x & 31;
  const int ln    = lane & 15;
  const int half1 = (lane >= 16);
  const int hi8   = half1 ? 8 : 0;
  const int hi16  = half1 ? 16 : 0;
  const int q0 = blockIdx.x * 16;
  const int bh = blockIdx.y;
  const bf16_t* Qh = Q + (size_t)bh * SEQ * HEAD_DIM;
  const bf16_t* Kh = K + (size_t)bh * SEQ * HEAD_DIM;
  const bf16_t* Vh = Vt + (size_t)bh * HEAD_DIM * SEQ;
  const int qidx = q0 + ln;

  v16bf qf[2];
#pragma unroll
  for (int dc = 0; dc < 2; ++dc)
    qf[dc] = *(const v16bf*)(Qh + (size_t)qidx * HEAD_DIM + dc * 32 + hi16);

  v8f o[4];
#pragma unroll
  for (int i = 0; i < 4; ++i) o[i] = {};
  float m_run = NEG_INF, l_run = 0.0f;

  for (int kb = 0; kb < q0 + 16; kb += 32) {  // 32 keys / iteration
    v8f s0 = {}, s1 = {};
#pragma unroll
    for (int dc = 0; dc < 2; ++dc) {
      v16bf k0 = load_frag_a(Kh + (size_t)(kb + ln) * HEAD_DIM + dc * 32, hi8);
      v16bf k1 = load_frag_a(Kh + (size_t)(kb + 16 + ln) * HEAD_DIM + dc * 32, hi8);
      s0 = wmma_bf16(k0, qf[dc], s0);
      s1 = wmma_bf16(k1, qf[dc], s1);
    }
    float p0[8], p1[8];
    float mt = NEG_INF;
#pragma unroll
    for (int r = 0; r < 8; ++r) {
      const int key0 = kb + r + hi8;
      const float a0 = (key0 <= qidx) ? s0[r] : NEG_INF;
      const float a1 = (key0 + 16 <= qidx) ? s1[r] : NEG_INF;
      p0[r] = a0;
      p1[r] = a1;
      mt = fmaxf(mt, fmaxf(a0, a1));
    }
    mt = fmaxf(mt, __shfl_xor(mt, 16, 32));
    const float m_new = fmaxf(m_run, mt);
    const float alpha = exp2f(m_run - m_new);  // first iter: exp2(-inf) = 0
    float rs = 0.0f;
#pragma unroll
    for (int r = 0; r < 8; ++r) {
      p0[r] = exp2f(p0[r] - m_new);
      p1[r] = exp2f(p1[r] - m_new);
      rs += p0[r] + p1[r];
    }
    rs += __shfl_xor(rs, 16, 32);
    l_run = l_run * alpha + rs;
    m_run = m_new;
#pragma unroll
    for (int i = 0; i < 4; ++i)
#pragma unroll
      for (int r = 0; r < 8; ++r) o[i][r] *= alpha;

    // Rebuild P^T as a B-fragment (lane = query column, K = 32 keys):
    //  half0 lane: keys 0..7 own s0, keys 8..15 from lane+16's s0
    //  half1 lane: keys 16..23 from lane-16's s1, keys 24..31 own s1
    v16bf pb;
#pragma unroll
    for (int r = 0; r < 8; ++r) {
      const float x0 = __shfl_xor(p0[r], 16, 32);
      const float x1 = __shfl_xor(p1[r], 16, 32);
      pb[r]     = f2bf(half1 ? x1 : p0[r]);
      pb[8 + r] = f2bf(half1 ? p1[r] : x0);
    }
#pragma unroll
    for (int i = 0; i < 4; ++i) {
      v16bf vf = load_frag_a(Vh + (size_t)(i * 16 + ln) * SEQ + kb, hi8);
      o[i] = wmma_bf16(vf, pb, o[i]);
    }
  }

  const float inv = 1.0f / l_run;
  const int b = bh >> 4, h = bh & 15;
  bf16_t* yrow = Y + (size_t)(b * SEQ + qidx) * D_MODEL + h * HEAD_DIM;
#pragma unroll
  for (int i = 0; i < 4; ++i) {
    v8bf pk;
#pragma unroll
    for (int r = 0; r < 8; ++r) pk[r] = f2bf(o[i][r] * inv);
    *(v8bf*)(yrow + i * 16 + hi8) = pk;  // d = i*16 + hi8 + r, contiguous
  }
}

// ---------------------------------------------------------------------------
// Host-side orchestration (graph-capture safe: only kernel launches on stream)
// Workspace layout: ~48 MB.
// ---------------------------------------------------------------------------
extern "C" void kernel_launch(void* const* d_in, const int* in_sizes, int n_in,
                              void* d_out, int out_size, void* d_ws,
                              size_t ws_size, hipStream_t stream) {
  (void)in_sizes; (void)n_in; (void)out_size; (void)ws_size;
  const float* x  = (const float*)d_in[0];
  const float* Wq = (const float*)d_in[1];
  const float* bq = (const float*)d_in[2];
  const float* Wk = (const float*)d_in[3];
  const float* bk = (const float*)d_in[4];
  const float* Wv = (const float*)d_in[5];
  const float* bv = (const float*)d_in[6];
  const float* Wo = (const float*)d_in[7];
  const float* bo = (const float*)d_in[8];

  char* ws = (char*)d_ws;
  const size_t MB = 1024u * 1024u;
  bf16_t* xb  = (bf16_t*)(ws + 0 * MB);   // 8 MB  [4096,1024] bf16
  bf16_t* wqb = (bf16_t*)(ws + 8 * MB);   // 2 MB
  bf16_t* wkb = (bf16_t*)(ws + 10 * MB);  // 2 MB
  bf16_t* wvb = (bf16_t*)(ws + 12 * MB);  // 2 MB
  bf16_t* wob = (bf16_t*)(ws + 14 * MB);  // 2 MB
  bf16_t* Qb  = (bf16_t*)(ws + 16 * MB);  // 8 MB  [B,H,T,Dh] (pre-scaled)
  bf16_t* Kb  = (bf16_t*)(ws + 24 * MB);  // 8 MB  [B,H,T,Dh]
  bf16_t* Vtb = (bf16_t*)(ws + 32 * MB);  // 8 MB  [B,H,Dh,T]
  bf16_t* Yb  = (bf16_t*)(ws + 40 * MB);  // 8 MB  [4096,1024] bf16

  // 1) downconvert activations + weights to bf16
  f32_to_bf16_kernel<<<2048, 256, 0, stream>>>(x, xb, MROWS * D_MODEL);
  f32_to_bf16_kernel<<<512, 256, 0, stream>>>(Wq, wqb, D_MODEL * D_MODEL);
  f32_to_bf16_kernel<<<512, 256, 0, stream>>>(Wk, wkb, D_MODEL * D_MODEL);
  f32_to_bf16_kernel<<<512, 256, 0, stream>>>(Wv, wvb, D_MODEL * D_MODEL);
  f32_to_bf16_kernel<<<512, 256, 0, stream>>>(Wo, wob, D_MODEL * D_MODEL);

  // 2) Q/K projections -> [B,H,T,Dh] bf16 (Q pre-scaled by 0.125*log2e)
  gemm_xwT_t<0><<<dim3(MROWS / 32, D_MODEL / 256), 128, 0, stream>>>(
      xb, wqb, bq, QSCALE, (void*)Qb);
  gemm_xwT_t<0><<<dim3(MROWS / 32, D_MODEL / 256), 128, 0, stream>>>(
      xb, wkb, bk, 1.0f, (void*)Kb);

  // 3) V projection -> transposed [B,H,Dh,T] bf16
  gemm_xwT_vT<<<dim3(MROWS / 256, D_MODEL / 32), 128, 0, stream>>>(
      xb, wvb, bv, Vtb);

  // 4) causal flash attention -> y bf16 [B,T,D]
  flash_attn_kernel<<<dim3(SEQ / 16, BATCH * N_HEADS), 32, 0, stream>>>(
      Qb, Kb, Vtb, Yb);

  // 5) output projection -> fp32 d_out [B,T,D]
  gemm_xwT_t<2><<<dim3(MROWS / 32, D_MODEL / 256), 128, 0, stream>>>(
      Yb, wob, bo, 1.0f, d_out);
}
